// Model3_77412490543855
// MI455X (gfx1250) — matrix operand order repaired and checked by reading, not compiled, not run on hardware
//
#include <hip/hip_runtime.h>
#include <hip/hip_bf16.h>
#include <math.h>

// ---------------------------------------------------------------------------
// CDNA5 (gfx1250) implementation of the tetris-policy transformer reference.
// f32 in memory, bf16 WMMA (v_wmma_f32_16x16x32_bf16) for all GEMMs, f32
// accumulate.  GEMM: 64x64 block tile, 4 waves, each wave = 2x2 WMMA tiles
// (4 wmma per K-step, 2x LDS operand reuse).  Fast path uses vectorized
// b128 global loads + packed b128 LDS stores; edge path uses clamped
// unconditional loads (no exec-mask ladders).
// ---------------------------------------------------------------------------

typedef __attribute__((ext_vector_type(16))) __bf16 v16bf;
typedef __attribute__((ext_vector_type(8)))  __bf16 v8bf;
typedef __attribute__((ext_vector_type(8)))  float  v8f;

static constexpr int Bsz  = 128;
static constexpr int Dm   = 256;
static constexpr int FFd  = 1024;
static constexpr int NH   = 4;
static constexpr int SEQn = 200;
static constexpr int L2n  = 56;

#define DEVI __device__ __forceinline__

DEVI unsigned short f2bf(float f) {            // round-to-nearest-even f32->bf16
  unsigned int u = __float_as_uint(f);
  u += 0x7FFFu + ((u >> 16) & 1u);
  return (unsigned short)(u >> 16);
}
DEVI unsigned int pack2(float lo, float hi) {
  return (unsigned int)f2bf(lo) | ((unsigned int)f2bf(hi) << 16);
}
DEVI float relu6f(float v) { return fminf(fmaxf(v, 0.f), 6.f); }

// ---------------------------------------------------------------------------
// Generic tiled WMMA GEMM:  C[M,N] = act( alpha * (A @ B) + bias )
// A: [M,K] row-major (lda).  B: transB ? [N,K] (ldb) : [K,N] (ldb).
// Batch: blockIdx.z = z1*nb2 + z2; offsets sX1*z1 + sX2*z2.
// 128 threads = 4 waves; block tile 64x64; wave tile 32x32 = 2x2 WMMA tiles.
// ---------------------------------------------------------------------------
#define TPAD 40   // LDS row pitch in bf16 elems (80 B -> 16B-aligned rows)

__global__ __launch_bounds__(128)
void gemm_kernel(const float* __restrict__ A, const float* __restrict__ Bm,
                 const float* __restrict__ bias, float* __restrict__ C,
                 int M, int N, int K, int lda, int ldb, int ldc,
                 long long sA1, long long sA2, long long sB1, long long sB2,
                 long long sC1, long long sC2, int nb2,
                 int transB, int act, float alpha)
{
  __shared__ alignas(16) unsigned short As[64 * TPAD];
  __shared__ alignas(16) unsigned short Bs[64 * TPAD];

  const int z1 = blockIdx.z / nb2, z2 = blockIdx.z % nb2;
  A  += z1 * sA1 + z2 * sA2;
  Bm += z1 * sB1 + z2 * sB2;
  C  += z1 * sC1 + z2 * sC2;

  const int n0 = blockIdx.x * 64, m0 = blockIdx.y * 64;
  const int t = threadIdx.x;
  const int lane = t & 31, wave = t >> 5;
  const int wm2 = (wave >> 1) * 32, wn2 = (wave & 1) * 32;
  const int lr = lane & 15, kh = lane >> 4;

  // uniform fast-path decision (per block)
  const bool fast = (m0 + 64 <= M) && (n0 + 64 <= N) && ((K & 31) == 0) &&
                    ((lda & 3) == 0) && ((ldb & 3) == 0);

  v8f a00 = {}, a01 = {}, a10 = {}, a11 = {};

  for (int k0 = 0; k0 < K; k0 += 32) {
    if (fast) {
      // ---- A: row = t>>1, 16 contiguous k; 4x b128 loads, 2x b128 stores ----
      {
        const int row = t >> 1, kb = (t & 1) * 16;
        const float4* src =
            reinterpret_cast<const float4*>(A + (long long)(m0 + row) * lda + k0 + kb);
        unsigned int* dst = reinterpret_cast<unsigned int*>(&As[row * TPAD + kb]);
        float4 f0 = src[0], f1 = src[1], f2 = src[2], f3 = src[3];
        uint4 p0, p1;
        p0.x = pack2(f0.x, f0.y); p0.y = pack2(f0.z, f0.w);
        p0.z = pack2(f1.x, f1.y); p0.w = pack2(f1.z, f1.w);
        p1.x = pack2(f2.x, f2.y); p1.y = pack2(f2.z, f2.w);
        p1.z = pack2(f3.x, f3.y); p1.w = pack2(f3.z, f3.w);
        reinterpret_cast<uint4*>(dst)[0] = p0;
        reinterpret_cast<uint4*>(dst)[1] = p1;
      }
      // ---- B ----
      if (transB) {            // B[N,K]: same pattern as A (rows are n)
        const int row = t >> 1, kb = (t & 1) * 16;
        const float4* src =
            reinterpret_cast<const float4*>(Bm + (long long)(n0 + row) * ldb + k0 + kb);
        unsigned int* dst = reinterpret_cast<unsigned int*>(&Bs[row * TPAD + kb]);
        float4 f0 = src[0], f1 = src[1], f2 = src[2], f3 = src[3];
        uint4 p0, p1;
        p0.x = pack2(f0.x, f0.y); p0.y = pack2(f0.z, f0.w);
        p0.z = pack2(f1.x, f1.y); p0.w = pack2(f1.z, f1.w);
        p1.x = pack2(f2.x, f2.y); p1.y = pack2(f2.z, f2.w);
        p1.z = pack2(f3.x, f3.y); p1.w = pack2(f3.z, f3.w);
        reinterpret_cast<uint4*>(dst)[0] = p0;
        reinterpret_cast<uint4*>(dst)[1] = p1;
      } else {                 // B[K,N]: read 16 contiguous n at row k, scatter
        const int k = t >> 2, nb = (t & 3) * 16;
        const float4* src =
            reinterpret_cast<const float4*>(Bm + (long long)(k0 + k) * ldb + n0 + nb);
#pragma unroll
        for (int s = 0; s < 4; ++s) {
          float4 f = src[s];
          Bs[(nb + 4 * s + 0) * TPAD + k] = f2bf(f.x);
          Bs[(nb + 4 * s + 1) * TPAD + k] = f2bf(f.y);
          Bs[(nb + 4 * s + 2) * TPAD + k] = f2bf(f.z);
          Bs[(nb + 4 * s + 3) * TPAD + k] = f2bf(f.w);
        }
      }
    } else {
      // ---- edge path: clamped unconditional loads + select-0 ----
#pragma unroll 4
      for (int idx = t; idx < 64 * 32; idx += 128) {
        const int row = idx >> 5, kk = idx & 31;
        const int gk = k0 + kk;
        const int kc = gk < K ? gk : K - 1;
        {
          const int gm = m0 + row;
          const int mc = gm < M ? gm : M - 1;
          float va = A[(long long)mc * lda + kc];
          As[row * TPAD + kk] = f2bf((gm < M && gk < K) ? va : 0.f);
        }
        {
          const int gn = n0 + row;
          const int nc = gn < N ? gn : N - 1;
          float vb = transB ? Bm[(long long)nc * ldb + kc]
                            : Bm[(long long)kc * ldb + nc];
          Bs[row * TPAD + kk] = f2bf((gn < N && gk < K) ? vb : 0.f);
        }
      }
    }
    __syncthreads();

    // ---- fragments (ISA 16-bit layouts) + 4 WMMAs ----
    {
      v16bf af0, af1, bf0, bf1;
      {
        const unsigned short* ap = &As[(wm2 + lr) * TPAD];
        v8bf x0 = *reinterpret_cast<const v8bf*>(ap + kh * 8);
        v8bf x1 = *reinterpret_cast<const v8bf*>(ap + 16 + kh * 8);
        const unsigned short* ap2 = &As[(wm2 + 16 + lr) * TPAD];
        v8bf y0 = *reinterpret_cast<const v8bf*>(ap2 + kh * 8);
        v8bf y1 = *reinterpret_cast<const v8bf*>(ap2 + 16 + kh * 8);
        const unsigned short* bp = &Bs[(wn2 + lr) * TPAD];
        v8bf z0 = *reinterpret_cast<const v8bf*>(bp + kh * 8);
        v8bf z1 = *reinterpret_cast<const v8bf*>(bp + 16 + kh * 8);
        const unsigned short* bp2 = &Bs[(wn2 + 16 + lr) * TPAD];
        v8bf w0 = *reinterpret_cast<const v8bf*>(bp2 + kh * 8);
        v8bf w1 = *reinterpret_cast<const v8bf*>(bp2 + 16 + kh * 8);
#pragma unroll
        for (int e = 0; e < 8; ++e) {
          af0[e] = x0[e]; af0[e + 8] = x1[e];
          af1[e] = y0[e]; af1[e + 8] = y1[e];
          bf0[e] = z0[e]; bf0[e + 8] = z1[e];
          bf1[e] = w0[e]; bf1[e + 8] = w1[e];
        }
      }
      a00 = __builtin_amdgcn_wmma_f32_16x16x32_bf16(false, af0, false, bf0, (short)0, a00, false, false);
      a01 = __builtin_amdgcn_wmma_f32_16x16x32_bf16(false, af0, false, bf1, (short)0, a01, false, false);
      a10 = __builtin_amdgcn_wmma_f32_16x16x32_bf16(false, af1, false, bf0, (short)0, a10, false, false);
      a11 = __builtin_amdgcn_wmma_f32_16x16x32_bf16(false, af1, false, bf1, (short)0, a11, false, false);
    }
    __syncthreads();
  }

  // ---- epilogue: VGPR r -> M = r + 8*kh, N = lane&15 ----
#pragma unroll
  for (int i = 0; i < 2; ++i) {
#pragma unroll
    for (int j = 0; j < 2; ++j) {
      const v8f& acc = (i == 0) ? (j == 0 ? a00 : a01) : (j == 0 ? a10 : a11);
      const int gn = n0 + wn2 + 16 * j + lr;
#pragma unroll
      for (int r = 0; r < 8; ++r) {
        const int gm = m0 + wm2 + 16 * i + r + 8 * kh;
        if (gm < M && gn < N) {
          float v = acc[r] * alpha;
          if (bias) v += bias[gn];
          if (act == 1) v = fmaxf(v, 0.f);
          else if (act == 2) v = relu6f(v);
          C[(long long)gm * ldc + gn] = v;
        }
      }
    }
  }
}

// ---------------------------------------------------------------------------
// Fused residual + LayerNorm (+ optional relu6).  One row per block.
// ---------------------------------------------------------------------------
__global__ __launch_bounds__(128)
void ln_kernel(const float* __restrict__ x, const float* __restrict__ res,
               const float* __restrict__ g, const float* __restrict__ b,
               float* __restrict__ y, int dim, int act)
{
  __shared__ float red[128];
  const long long row = blockIdx.x;
  const float* xr = x + row * dim;
  const float* rr = res ? res + row * dim : nullptr;
  const int t = threadIdx.x;

  float s = 0.f, ss = 0.f;
  for (int i = t; i < dim; i += 128) {
    float v = xr[i] + (rr ? rr[i] : 0.f);
    s += v; ss += v * v;
  }
  red[t] = s; __syncthreads();
  for (int o = 64; o > 0; o >>= 1) { if (t < o) red[t] += red[t + o]; __syncthreads(); }
  const float mean = red[0] / dim;
  __syncthreads();
  red[t] = ss; __syncthreads();
  for (int o = 64; o > 0; o >>= 1) { if (t < o) red[t] += red[t + o]; __syncthreads(); }
  const float var = red[0] / dim - mean * mean;
  const float inv = rsqrtf(var + 1e-5f);

  for (int i = t; i < dim; i += 128) {
    float v = xr[i] + (rr ? rr[i] : 0.f);
    float o = (v - mean) * inv * g[i] + b[i];
    if (act == 2) o = relu6f(o);
    y[row * dim + i] = o;
  }
}

// ---------------------------------------------------------------------------
// Softmax over last dim; optional additive bias with (b,q,n)->offset strides.
// ---------------------------------------------------------------------------
__global__ __launch_bounds__(64)
void softmax_kernel(float* __restrict__ s, const float* __restrict__ bias,
                    int len, int Q, int Nh,
                    long long sb_b, long long sb_n, long long sb_q)
{
  __shared__ float red[64];
  const long long r = blockIdx.x;
  float* p = s + r * len;
  const float* bp = nullptr;
  if (bias) {
    long long qb = (long long)Q * Nh;
    long long bb = r / qb, rem = r % qb;
    long long q = rem / Nh, n = rem % Nh;
    bp = bias + bb * sb_b + n * sb_n + q * sb_q;
  }
  const int t = threadIdx.x;
  float mx = -1e30f;
  for (int i = t; i < len; i += 64) mx = fmaxf(mx, p[i] + (bp ? bp[i] : 0.f));
  red[t] = mx; __syncthreads();
  for (int o = 32; o > 0; o >>= 1) { if (t < o) red[t] = fmaxf(red[t], red[t + o]); __syncthreads(); }
  mx = red[0]; __syncthreads();
  float sum = 0.f;
  for (int i = t; i < len; i += 64) {
    float v = __expf(p[i] + (bp ? bp[i] : 0.f) - mx);
    p[i] = v; sum += v;
  }
  red[t] = sum; __syncthreads();
  for (int o = 32; o > 0; o >>= 1) { if (t < o) red[t] += red[t + o]; __syncthreads(); }
  const float invs = 1.f / red[0];
  for (int i = t; i < len; i += 64) p[i] *= invs;
}

// --------------------------- small elementwise kernels ----------------------
static constexpr float LOG1E4 = 9.210340371976184f;

__global__ void pos_kernel(const float* __restrict__ xyrot, float* __restrict__ pos, int total) {
  int i = blockIdx.x * 256 + threadIdx.x; if (i >= total) return;
  int c = i & 3, tt = (i >> 2) % 14, b = i / 56;
  pos[i] = (tt == 0) ? xyrot[b * 4 + c] : -1.f;
}

__global__ void pp_kernel(const float* __restrict__ emb, const int* __restrict__ queue,
                          const int* __restrict__ remains, const float* __restrict__ loc2,
                          float* __restrict__ pp, int total) {
  int i = blockIdx.x * 256 + threadIdx.x; if (i >= total) return;
  int d = i & 255, tt = (i >> 8) % 14, b = i / (14 * 256);
  int idx = (tt < 7) ? queue[b * 7 + tt] : remains[b * 7 + tt - 7];
  int j = d >> 1;
  float div = __expf(-(float)(2 * j) * LOG1E4 / 256.f);
  float arg = ((tt < 7) ? (float)tt : 7.f) * div;
  float pe = (d & 1) ? __cosf(arg) : __sinf(arg);
  pp[i] = emb[idx * 256 + d] + loc2[i] + pe;
}

__global__ void dw1_kernel(const float* __restrict__ board, const float* __restrict__ w,
                           const float* __restrict__ bb, float* __restrict__ out) {
  int b = blockIdx.x, t = threadIdx.x; if (t >= 200) return;
  int y = t / 10, x = t % 10;
  const float* bd = board + (long long)b * 22 * 12;
  float s = bb[0];
  for (int i = 0; i < 3; ++i)
    for (int j = 0; j < 3; ++j) s += bd[(y + i) * 12 + (x + j)] * w[i * 3 + j];
  out[(long long)b * 200 + t] = s;
}

__global__ void pw1_kernel(const float* __restrict__ x, const float* __restrict__ w,
                           const float* __restrict__ bias, const float* __restrict__ g,
                           const float* __restrict__ bnb, float* __restrict__ out, int total) {
  int i = blockIdx.x * 256 + threadIdx.x; if (i >= total) return;
  int p = i % 200, c = (i / 200) & 127, b = i / (200 * 128);
  float geff = g[c] * rsqrtf(1.f + 1e-5f);
  float v = x[(long long)b * 200 + p] * w[c] + bias[c];
  out[i] = relu6f(v * geff + bnb[c]);
}

__global__ void dw2_kernel(const float* __restrict__ x, const float* __restrict__ w,
                           const float* __restrict__ bias, float* __restrict__ out, int total) {
  int i = blockIdx.x * 256 + threadIdx.x; if (i >= total) return;
  int xx = i % 10, yy = (i / 10) % 20, c = (i / 200) & 127, b = i / (200 * 128);
  const float* xp = x + ((long long)b * 128 + c) * 200;
  const float* wp = w + c * 9;
  float s = bias[c];
  for (int dy = -1; dy <= 1; ++dy)
    for (int dx = -1; dx <= 1; ++dx) {
      int Y = yy + dy, X = xx + dx;
      if (Y >= 0 && Y < 20 && X >= 0 && X < 10) s += xp[Y * 10 + X] * wp[(dy + 1) * 3 + dx + 1];
    }
  out[i] = s;
}

__global__ __launch_bounds__(256)
void pw2_kernel(const float* __restrict__ x, const float* __restrict__ w,
                const float* __restrict__ bias, const float* __restrict__ g,
                const float* __restrict__ bnb, float* __restrict__ bwp) {
  __shared__ float xs[128];
  int bp = blockIdx.x, b = bp / 200, p = bp % 200;
  int d = threadIdx.x;
  if (d < 128) xs[d] = x[((long long)b * 128 + d) * 200 + p];
  __syncthreads();
  float s = bias[d];
  const float* wp = w + d * 128;
  for (int ic = 0; ic < 128; ++ic) s += xs[ic] * wp[ic];
  s = relu6f(s * g[d] * rsqrtf(1.f + 1e-5f) + bnb[d]);
  int y = p / 10, xc = p % 10, j = d >> 1;
  float ex = __expf(-(float)(4 * j) * LOG1E4 / 256.f);
  float pe = (d & 1) ? __cosf(xc * ex) : __sinf(y * ex);
  bwp[((long long)b * 200 + p) * 256 + d] = s + pe;
}

__global__ void zcat_kernel(const float* __restrict__ bwp, const float* __restrict__ pwr,
                            float* __restrict__ z, int total) {
  int i = blockIdx.x * 256 + threadIdx.x; if (i >= total) return;
  int t = i & 255, d = (i >> 8) & 255, b = i >> 16;
  z[i] = (t < 200) ? bwp[((long long)b * 200 + t) * 256 + d]
                   : pwr[((long long)b * 56 + (t - 200)) * 256 + d];
}

__global__ void zflat_kernel(const float* __restrict__ z1, float* __restrict__ zf, int total) {
  int i = blockIdx.x * 256 + threadIdx.x; if (i >= total) return;
  int f = i & 8191, b = i >> 13;
  int j = f >> 8, d = f & 255;
  zf[i] = relu6f(z1[((long long)b * 256 + d) * 32 + j]);
}

__global__ void copytok0_kernel(const float* __restrict__ pp, float* __restrict__ x, int total) {
  int i = blockIdx.x * 256 + threadIdx.x; if (i >= total) return;
  int b = i >> 8, d = i & 255;
  x[i] = pp[(long long)b * 14 * 256 + d];
}

// --------------------------- param structs ---------------------------------
struct MHAW { const float *qw,*qb,*kw,*kb,*vw,*vb,*ow,*ob; };
struct DecW { MHAW sa, ca; const float *l1g,*l1b,*l2g,*l2b,*l3g,*l3b,*f1w,*f1b,*f2w,*f2b; };

// ---------------------------------------------------------------------------
extern "C" void kernel_launch(void* const* d_in, const int* in_sizes, int n_in,
                              void* d_out, int out_size, void* d_ws, size_t ws_size,
                              hipStream_t stream)
{
  (void)in_sizes; (void)n_in; (void)out_size; (void)ws_size;
  const float* xyrot    = (const float*)d_in[0];
  const int*   queue    = (const int*)  d_in[1];
  const int*   remains  = (const int*)  d_in[2];
  const float* board    = (const float*)d_in[3];
  const float* state_in = (const float*)d_in[4];
  float* outp = (float*)d_out;

  // ---- walk param leaves (make_params insertion order) ----
  int pi = 5;
  auto nf = [&]() -> const float* { return (const float*)d_in[pi++]; };
  const float *loc1w=nf(),*loc1b=nf(),*loc2w=nf(),*loc2b=nf();
  const float *piece_emb=nf();
  const float *rotw=nf(),*rotb=nf();
  const float *dw1w=nf(),*dw1b=nf(),*pw1w=nf(),*pw1b=nf(),*bn1g=nf(),*bn1b=nf();
  const float *dw2w=nf(),*dw2b=nf(),*pw2w=nf(),*pw2b=nf(),*bn2g=nf(),*bn2b=nf();
  const float *a_qw=nf(),*a_qb=nf(),*a_kw=nf(),*a_kb=nf(),*a_vw=nf(),*a_vb=nf();
  const float *b_l1w=nf();
  const float *b_l2w=nf(),*b_l2b=nf(),*b_ln1g=nf(),*b_ln1b=nf();
  const float *b_l3w=nf(),*b_l3b=nf(),*b_ln2g=nf(),*b_ln2b=nf();
  const float *b_l4w=nf(),*b_l4b=nf();
  const float *n1g=nf(),*n1b=nf(),*n2g=nf(),*n2b=nf();
  const float *aff1w=nf(),*aff1b=nf(),*aff2w=nf(),*aff2b=nf();
  auto rdMha = [&](MHAW& m){ m.qw=nf();m.qb=nf();m.kw=nf();m.kb=nf();m.vw=nf();m.vb=nf();m.ow=nf();m.ob=nf(); };
  auto rdDec = [&](DecW& d){ rdMha(d.sa); rdMha(d.ca);
    d.l1g=nf();d.l1b=nf(); d.l2g=nf();d.l2b=nf(); d.l3g=nf();d.l3b=nf();
    d.f1w=nf();d.f1b=nf(); d.f2w=nf();d.f2b=nf(); };
  DecW macroL[2], bdecL[2], moveL[2];
  rdDec(macroL[0]); rdDec(macroL[1]);
  rdDec(bdecL[0]);  rdDec(bdecL[1]);
  rdDec(moveL[0]);  rdDec(moveL[1]);
  const float *fin1w=nf(),*fin1b=nf(),*fin2w=nf(),*fin2b=nf();
  const float *fng=nf(),*fnb=nf();
  const float *pol1w=nf(),*pol1b=nf(),*pol2w=nf(),*pol2b=nf();
  const float *valw=nf(),*valb=nf();

  // ---- workspace layout (phase-overlaid bump allocator) ----
  float* base = (float*)d_ws;
  size_t off = 0;
  auto alloc = [&](size_t n) -> float* { float* p = base + off; off += (n + 63) & ~(size_t)63; return p; };
  float* pp    = alloc((size_t)Bsz*14*256);
  float* pwr   = alloc((size_t)Bsz*56*256);
  float* bwp   = alloc((size_t)Bsz*200*256);
  float* biasT = alloc((size_t)Bsz*4*11200);
  float* s0    = alloc((size_t)Bsz*200*256);
  float* s1    = alloc((size_t)Bsz*200*256);
  float* t_a   = alloc((size_t)Bsz*200*256);
  float* t_b   = alloc((size_t)Bsz*200*256);
  float* xq0   = alloc((size_t)Bsz*256);
  float* m0b   = alloc((size_t)Bsz*256);
  float* m1b   = alloc((size_t)Bsz*256);
  float* f1o   = alloc((size_t)Bsz*1024);
  float* f2o   = alloc((size_t)Bsz*1024);
  float* xh    = alloc((size_t)Bsz*1024);
  float* p1o   = alloc((size_t)Bsz*1024);
  const size_t mark = off;
  float* posb  = alloc((size_t)Bsz*14*4);
  float* locb  = alloc((size_t)Bsz*14*256);
  float* loc2v = alloc((size_t)Bsz*14*256);
  off = mark;
  float* t1 = alloc((size_t)Bsz*200);
  float* t2 = alloc((size_t)Bsz*128*200);
  float* t3 = alloc((size_t)Bsz*128*200);
  off = mark;
  float* zcat = alloc((size_t)Bsz*256*256);
  float* z1   = alloc((size_t)Bsz*256*32);
  float* zfl  = alloc((size_t)Bsz*8192);
  float* z2   = alloc((size_t)Bsz*128);
  float* z2n  = alloc((size_t)Bsz*128);
  float* z3   = alloc((size_t)Bsz*512);
  float* z3n  = alloc((size_t)Bsz*512);
  off = mark;
  float* qb_  = alloc((size_t)Bsz*200*256);
  float* kb_  = alloc((size_t)Bsz*200*256);
  float* vb_  = alloc((size_t)Bsz*200*256);
  float* ob_  = alloc((size_t)Bsz*200*256);
  float* ob2  = alloc((size_t)Bsz*200*256);
  float* ffo  = alloc((size_t)Bsz*200*256);
  float* hbuf = alloc((size_t)Bsz*200*1024);   // overlays scores (bigger)
  float* sc   = hbuf;                          // probs dead before h written

  // ---- helpers ----
  auto gemm = [&](const float* A, const float* Bm, const float* bias, float* C,
                  int M, int N, int K, int lda, int ldb, int ldc,
                  long long sA1, long long sA2, long long sB1, long long sB2,
                  long long sC1, long long sC2, int nb1, int nb2,
                  int transB, int act, float alpha) {
    dim3 g((N + 63) / 64, (M + 63) / 64, nb1 * nb2);
    gemm_kernel<<<g, dim3(128), 0, stream>>>(A, Bm, bias, C, M, N, K, lda, ldb, ldc,
                                             sA1, sA2, sB1, sB2, sC1, sC2, nb2,
                                             transB, act, alpha);
  };
  auto lin = [&](const float* A, const float* W, const float* b, float* C,
                 int M, int K, int N, int act) {
    gemm(A, W, b, C, M, N, K, K, N, N, 0,0,0,0,0,0, 1,1, 0, act, 1.f);
  };
  auto lnL = [&](const float* x, const float* res, const float* g, const float* b,
                 float* y, long long rows, int dim, int act) {
    ln_kernel<<<dim3((unsigned)rows), dim3(128), 0, stream>>>(x, res, g, b, y, dim, act);
  };
  auto mha = [&](const float* xq, int lq, const float* xkv, int lk, const MHAW& w) {
    lin(xq,  w.qw, w.qb, qb_, Bsz*lq, 256, 256, 0);
    lin(xkv, w.kw, w.kb, kb_, Bsz*lk, 256, 256, 0);
    lin(xkv, w.vw, w.vb, vb_, Bsz*lk, 256, 256, 0);
    gemm(qb_, kb_, nullptr, sc, lq, lk, 64, 256, 256, lk,
         (long long)lq*256, 64, (long long)lk*256, 64,
         (long long)4*lq*lk, (long long)lq*lk, Bsz, 4, 1, 0, 0.125f);
    softmax_kernel<<<dim3((unsigned)(Bsz*4*lq)), dim3(64), 0, stream>>>(
        sc, nullptr, lk, 1, 1, 0LL, 0LL, 0LL);
    gemm(sc, vb_, nullptr, ob_, lq, 64, lk, lk, 256, 256,
         (long long)4*lq*lk, (long long)lq*lk, (long long)lk*256, 64,
         (long long)lq*256, 64, Bsz, 4, 0, 0, 1.f);
    lin(ob_, w.ow, w.ob, ob2, Bsz*lq, 256, 256, 0);
  };
  auto run_dec = [&](const float* xin, int lq, const float* mem, int lk,
                     const DecW& w, float* out) {
    mha(xin, lq, xin, lq, w.sa);
    lnL(ob2, xin, w.l1g, w.l1b, t_a, (long long)Bsz*lq, 256, 0);
    mha(t_a, lq, mem, lk, w.ca);
    lnL(ob2, t_a, w.l2g, w.l2b, t_b, (long long)Bsz*lq, 256, 0);
    lin(t_b, w.f1w, w.f1b, hbuf, Bsz*lq, 256, 1024, 1);
    lin(hbuf, w.f2w, w.f2b, ffo, Bsz*lq, 1024, 256, 0);
    lnL(ffo, t_b, w.l3g, w.l3b, out, (long long)Bsz*lq, 256, 0);
  };
  auto blk = [&](int tot) { return dim3((unsigned)((tot + 255) / 256)); };

  // ================= phase 1: piece pipeline =================
  int tot = Bsz*14*4;
  pos_kernel<<<blk(tot), 256, 0, stream>>>(xyrot, posb, tot);
  lin(posb, loc1w, loc1b, locb, Bsz*14, 4, 256, 2);
  lin(locb, loc2w, loc2b, loc2v, Bsz*14, 256, 256, 0);
  tot = Bsz*14*256;
  pp_kernel<<<blk(tot), 256, 0, stream>>>(piece_emb, queue, remains, loc2v, pp, tot);
  lin(pp, rotw, rotb, pwr, Bsz*14, 256, 1024, 2);

  // ================= phase 2: conv stem =================
  dw1_kernel<<<dim3(Bsz), 256, 0, stream>>>(board, dw1w, dw1b, t1);
  tot = Bsz*128*200;
  pw1_kernel<<<blk(tot), 256, 0, stream>>>(t1, pw1w, pw1b, bn1g, bn1b, t2, tot);
  dw2_kernel<<<blk(tot), 256, 0, stream>>>(t2, dw2w, dw2b, t3, tot);
  pw2_kernel<<<dim3(Bsz*200), 256, 0, stream>>>(t3, pw2w, pw2b, bn2g, bn2b, bwp);

  // ================= phase 3: bias net =================
  tot = Bsz*256*256;
  zcat_kernel<<<blk(tot), 256, 0, stream>>>(bwp, pwr, zcat, tot);
  gemm(zcat, b_l1w, nullptr, z1, 256, 32, 256, 256, 32, 32,
       65536, 0, 0, 0, 8192, 0, Bsz, 1, 0, 0, 1.f);
  tot = Bsz*8192;
  zflat_kernel<<<blk(tot), 256, 0, stream>>>(z1, zfl, tot);
  lin(zfl, b_l2w, b_l2b, z2, Bsz, 8192, 128, 0);
  lnL(z2, nullptr, b_ln1g, b_ln1b, z2n, Bsz, 128, 2);
  lin(z2n, b_l3w, b_l3b, z3, Bsz, 128, 512, 0);
  lnL(z3, nullptr, b_ln2g, b_ln2b, z3n, (long long)Bsz*4, 128, 2);
  lin(z3n, b_l4w, b_l4b, biasT, Bsz*4, 128, 11200, 0);

  // ================= attn_with_bias =================
  lin(bwp, a_qw, a_qb, qb_, Bsz*200, 256, 256, 0);
  lin(pwr, a_kw, a_kb, kb_, Bsz*56, 256, 256, 0);
  lin(pwr, a_vw, a_vb, vb_, Bsz*56, 256, 256, 0);
  gemm(qb_, kb_, nullptr, sc, 200, 56, 64, 256, 256, 224,
       51200, 64, 14336, 64, 44800, 56, Bsz, 4, 1, 0, 0.125f);
  softmax_kernel<<<dim3((unsigned)(Bsz*200*4)), dim3(64), 0, stream>>>(
      sc, biasT, 56, 200, 4, 44800LL, 11200LL, 56LL);
  gemm(sc, vb_, nullptr, ob_, 200, 64, 56, 224, 256, 256,
       44800, 56, 14336, 64, 51200, 64, Bsz, 4, 0, 0, 1.f);
  lnL(ob_, bwp, n1g, n1b, s0, (long long)Bsz*200, 256, 0);
  lin(s0, aff1w, aff1b, hbuf, Bsz*200, 256, 1024, 1);
  lin(hbuf, aff2w, aff2b, ffo, Bsz*200, 1024, 256, 0);
  lnL(ffo, s0, n2g, n2b, s1, (long long)Bsz*200, 256, 0);

  // ================= decoder stacks =================
  float* cur = s1; float* nxt = s0;
  for (int l = 0; l < 2; ++l) { run_dec(cur, 200, state_in, 200, macroL[l], nxt); float* t = cur; cur = nxt; nxt = t; }
  for (int l = 0; l < 2; ++l) { run_dec(cur, 200, pwr, 56, bdecL[l], nxt); float* t = cur; cur = nxt; nxt = t; }

  tot = Bsz*256;
  copytok0_kernel<<<blk(tot), 256, 0, stream>>>(pp, xq0, tot);
  run_dec(xq0, 1, cur, 200, moveL[0], m0b);
  run_dec(m0b, 1, cur, 200, moveL[1], m1b);

  // ================= head =================
  lin(m1b, fin1w, fin1b, f1o, Bsz, 256, 1024, 2);
  lin(f1o, fin2w, fin2b, f2o, Bsz, 1024, 1024, 0);
  lnL(f2o, f1o, fng, fnb, xh, Bsz, 1024, 0);
  lin(xh, pol1w, pol1b, p1o, Bsz, 1024, 1024, 2);
  lin(p1o, pol2w, pol2b, outp, Bsz, 1024, 10, 0);
  lin(xh, valw, valb, outp + Bsz*10, Bsz, 1024, 1024, 2);
}
